// OctreeResBlock2_8624294331036
// MI455X (gfx1250) — matrix operand order repaired and checked by reading, not compiled, not run on hardware
//
#include <hip/hip_runtime.h>
#include <hip/hip_bf16.h>
#include <cstdint>

// ---------------------------------------------------------------------------
// Types for CDNA5 WMMA (wave32): bf16 A/B fragments (16 bf16 = 8 VGPRs), f32 C/D.
// ---------------------------------------------------------------------------
typedef __attribute__((ext_vector_type(16))) __bf16 v16bf;
typedef __attribute__((ext_vector_type(8)))  float  v8f;
typedef int v4i_ __attribute__((vector_size(16)));   // async builtin's pointer element type

// ---------------------------------------------------------------------------
// gfx1250 async memory->LDS path (guarded; falls back to sync copy).
// ---------------------------------------------------------------------------
#if defined(__has_builtin)
# if __has_builtin(__builtin_amdgcn_global_load_async_to_lds_b128) && \
     __has_builtin(__builtin_amdgcn_s_wait_asynccnt)
#  define USE_ASYNC_LDS 1
# endif
#endif
#ifndef USE_ASYNC_LDS
# define USE_ASYNC_LDS 0
#endif

#if USE_ASYNC_LDS
# define ASYNC_CP16(gsrc, ldst)                                                \
    __builtin_amdgcn_global_load_async_to_lds_b128((v4i_*)(gsrc), (v4i_*)(ldst), 0, 0)
# define ASYNC_WAIT(n) __builtin_amdgcn_s_wait_asynccnt(n)
#else
# define ASYNC_WAIT(n)
#endif

// float -> bf16 round-to-nearest-even via bit manipulation (no __bf16 arithmetic).
__device__ __forceinline__ __bf16 f2bf(float f) {
    union { float f; uint32_t u; } a; a.f = f;
    uint32_t r = a.u + 0x7FFFu + ((a.u >> 16) & 1u);
    union { uint16_t u; __bf16 b; } o; o.u = (uint16_t)(r >> 16);
    return o.b;
}

// ---------------------------------------------------------------------------
// Gather-GEMM with WMMA:  out[o,n] = sum_{k<NTAPS} sum_{c<CIN} xT[idx(n,k)][c] * wT[k][o][c]
//   idx(n,k) = GATHER ? neigh[n*27+k] : n
// Block: 256 threads (8 waves). Tile: M=128 (all Cout) x NT=128 points.
// Wave (mw,nw): mw in 0..3 -> 32 M-rows, nw in 0..1 -> 64 N-cols
//   -> acc[2][4] of v_wmma_f32_16x16x32_bf16.
// LDS double-buffered (parity offset); tap k+1 staged with
// global_load_async_to_lds_b128 (ASYNCcnt) while tap k computes on the XDL pipe.
// ---------------------------------------------------------------------------
template<int CIN, int NTAPS, bool GATHER>
__global__ __launch_bounds__(256)
void conv_wmma(const __bf16* __restrict__ xT,    // [NPTS][CIN] bf16
               const __bf16* __restrict__ wT,    // [NTAPS][128][CIN] bf16
               const int*    __restrict__ neigh, // [NPTS][27] (GATHER only)
               float*        __restrict__ out,   // [128][NPTS] f32
               int NPTS)
{
    constexpr int NT   = 128;       // points per block
    constexpr int WROW = CIN + 8;   // bf16 units; +16B pad keeps b128 align, kills bank conflicts
    constexpr int XROW = CIN + 8;
    constexpr int VPR  = CIN / 8;   // uint4 (8 bf16) chunks per row
    constexpr int HALF = 128 * WROW + NT * XROW;   // one buffer (bf16 units)
    constexpr int ITW  = 128 * VPR / 256;          // weight chunks per thread per tap
    constexpr int ITX  = NT  * VPR / 256;          // activation chunks per thread per tap
    constexpr int PER_TAP = ITW + ITX;             // async ops per wave per tap

    extern __shared__ __bf16 smem[];

    const int tid  = threadIdx.x;
    __builtin_assume(tid < 256);
    const int wave = tid >> 5;
    const int lane = tid & 31;
    const int hl   = lane & 15;     // half-lane index
    const int hi   = lane >> 4;     // which 16-lane half
    const int mw   = wave & 3;      // 4 M-groups of 32 rows
    const int nw   = wave >> 2;     // 2 N-groups of 64 cols
    const int n0   = blockIdx.x * NT;

    v8f acc[2][4];
    const v8f vzero = {0.f,0.f,0.f,0.f,0.f,0.f,0.f,0.f};
    #pragma unroll
    for (int mt = 0; mt < 2; ++mt)
        #pragma unroll
        for (int nt = 0; nt < 4; ++nt) acc[mt][nt] = vzero;

    // ---- staging: tap k -> buffer parity p (counted loops: no exec guards) ----
    auto stage = [&](int k, int p) {
        __bf16* dw = smem + p * HALF;
        __bf16* dx = dw + 128 * WROW;
        const __bf16* wsrc = wT + (size_t)k * 128 * CIN;
        #pragma unroll
        for (int it = 0; it < ITW; ++it) {
            int c = tid + it * 256;
            int row = c / VPR, v = c % VPR;
#if USE_ASYNC_LDS
            ASYNC_CP16(wsrc + row * CIN + v * 8, dw + row * WROW + v * 8);
#else
            *(uint4*)(dw + row * WROW + v * 8) = *(const uint4*)(wsrc + row * CIN + v * 8);
#endif
        }
        #pragma unroll
        for (int it = 0; it < ITX; ++it) {
            int c = tid + it * 256;
            int row = c / VPR, v = c % VPR;
            int n   = n0 + row;
            int idx = GATHER ? neigh[n * 27 + k] : n;
#if USE_ASYNC_LDS
            ASYNC_CP16(xT + (size_t)idx * CIN + v * 8, dx + row * XROW + v * 8);
#else
            *(uint4*)(dx + row * XROW + v * 8) = *(const uint4*)(xT + (size_t)idx * CIN + v * 8);
#endif
        }
    };

    stage(0, 0);

    for (int k = 0; k < NTAPS; ++k) {
        if (k + 1 < NTAPS) {
            stage(k + 1, (k + 1) & 1);
            ASYNC_WAIT(PER_TAP);   // retire tap k's batch, keep tap k+1 in flight
        } else {
            ASYNC_WAIT(0);
        }
        __syncthreads();           // all waves' tap-k data resident in LDS

        const __bf16* lw = smem + (k & 1) * HALF;
        const __bf16* lx = lw + 128 * WROW;

        #pragma unroll
        for (int ck = 0; ck < CIN / 32; ++ck) {
            // A fragments (16x32 bf16): lane(hi,hl) holds row M=hl,
            // K = ck*32 + hi*8 + {0..7}  then  K = ck*32 + 16 + hi*8 + {0..7}.
            v16bf afrag[2];
            #pragma unroll
            for (int mt = 0; mt < 2; ++mt) {
                int row = mw * 32 + mt * 16 + hl;
                const __bf16* p = lw + row * WROW + ck * 32 + hi * 8;
                union { v16bf v; uint4 q[2]; } u;
                u.q[0] = *(const uint4*)(p);
                u.q[1] = *(const uint4*)(p + 16);
                afrag[mt] = u.v;
            }
            // B fragments (32x16 bf16): lane holds col N=hl, K = ck*32 + hi*8 + {0..7} then ck*32 + 16 + hi*8 + {0..7}.
            v16bf bfrag[4];
            #pragma unroll
            for (int nt = 0; nt < 4; ++nt) {
                int row = nw * 64 + nt * 16 + hl;
                const __bf16* p = lx + row * XROW + ck * 32 + hi * 8;
                union { v16bf v; uint4 q[2]; } u;
                u.q[0] = *(const uint4*)(p);
                u.q[1] = *(const uint4*)(p + 16);
                bfrag[nt] = u.v;
            }
            #pragma unroll
            for (int mt = 0; mt < 2; ++mt)
                #pragma unroll
                for (int nt = 0; nt < 4; ++nt)
                    acc[mt][nt] = __builtin_amdgcn_wmma_f32_16x16x32_bf16(
                        false, afrag[mt], false, bfrag[nt],
                        (short)0, acc[mt][nt], false, false);
        }
        __syncthreads();           // readers done before buffer is overwritten
    }

    // D layout: n = tileN + hl ; m = tileM + r + 8*hi  (VGPR r of 8)
    #pragma unroll
    for (int mt = 0; mt < 2; ++mt)
        #pragma unroll
        for (int nt = 0; nt < 4; ++nt) {
            int ncol  = n0 + nw * 64 + nt * 16 + hl;
            int mbase = mw * 32 + mt * 16 + hi * 8;
            #pragma unroll
            for (int r = 0; r < 8; ++r)
                out[(size_t)(mbase + r) * NPTS + ncol] = acc[mt][nt][r];
        }
}

// ---------------------------------------------------------------------------
// Tiled transpose: x [C][NPTS] f32 -> xT [NPTS][C] bf16
// ---------------------------------------------------------------------------
__global__ void transpose_to_bf16(const float* __restrict__ src, __bf16* __restrict__ dst,
                                  int NPTS, int C)
{
    __shared__ float t[32][33];
    int n0 = blockIdx.x * 32, c0 = blockIdx.y * 32;
    int tx = threadIdx.x, ty = threadIdx.y;            // 32 x 8
    #pragma unroll
    for (int r = 0; r < 32; r += 8)
        t[ty + r][tx] = src[(size_t)(c0 + ty + r) * NPTS + n0 + tx];
    __syncthreads();
    #pragma unroll
    for (int r = 0; r < 32; r += 8)
        dst[(size_t)(n0 + ty + r) * C + c0 + tx] = f2bf(t[tx][ty + r]);
}

// ---------------------------------------------------------------------------
// BN + ReLU + transpose: raw [128][NPTS] f32 -> c1T [NPTS][128] bf16
// stats: mean[0..127], rstd[128..255]
// ---------------------------------------------------------------------------
__global__ void bnrelu_transpose(const float* __restrict__ raw,
                                 const float* __restrict__ stats,
                                 const float* __restrict__ gamma,
                                 const float* __restrict__ beta,
                                 __bf16* __restrict__ dst, int NPTS)
{
    __shared__ float t[32][33];
    int n0 = blockIdx.x * 32, c0 = blockIdx.y * 32;
    int tx = threadIdx.x, ty = threadIdx.y;
    #pragma unroll
    for (int r = 0; r < 32; r += 8) {
        int c = c0 + ty + r;
        float v = raw[(size_t)c * NPTS + n0 + tx];
        v = gamma[c] * (v - stats[c]) * stats[128 + c] + beta[c];
        t[ty + r][tx] = fmaxf(v, 0.f);
    }
    __syncthreads();
    #pragma unroll
    for (int r = 0; r < 32; r += 8)
        dst[(size_t)(n0 + ty + r) * 128 + c0 + tx] = f2bf(t[tx][ty + r]);
}

// ---------------------------------------------------------------------------
// Weight repack: w [128][CIN][K] f32 -> wT [K][128][CIN] bf16
// ---------------------------------------------------------------------------
__global__ void wtrans(const float* __restrict__ w, __bf16* __restrict__ wt, int CIN, int K)
{
    int i = blockIdx.x * 256 + threadIdx.x;
    int total = 128 * CIN * K;
    if (i >= total) return;
    int c = i % CIN; int rem = i / CIN; int o = rem % 128; int k = rem / 128;
    wt[i] = f2bf(w[(o * CIN + c) * K + k]);
}

// ---------------------------------------------------------------------------
// Per-channel mean / rstd over N (deterministic tree reduction, block/channel)
// ---------------------------------------------------------------------------
__global__ __launch_bounds__(256)
void bn_stats(const float* __restrict__ raw, float* __restrict__ stats, int NPTS)
{
    __shared__ float ss[256], sq[256];
    int ch = blockIdx.x, tid = threadIdx.x;
    const float4* p = (const float4*)(raw + (size_t)ch * NPTS);
    float s = 0.f, q = 0.f;
    for (int i = tid; i < NPTS / 4; i += 256) {
        float4 v = p[i];
        s += (v.x + v.y) + (v.z + v.w);
        q += (v.x * v.x + v.y * v.y) + (v.z * v.z + v.w * v.w);
    }
    ss[tid] = s; sq[tid] = q;
    __syncthreads();
    for (int o = 128; o > 0; o >>= 1) {
        if (tid < o) { ss[tid] += ss[tid + o]; sq[tid] += sq[tid + o]; }
        __syncthreads();
    }
    if (tid == 0) {
        float m = ss[0] / (float)NPTS;
        float v = sq[0] / (float)NPTS - m * m;
        stats[ch]       = m;
        stats[128 + ch] = rsqrtf(v + 1e-3f);
    }
}

// ---------------------------------------------------------------------------
// out = relu( bn_b(c2) + bn_s(skip) )   (float4 vectorized)
// ---------------------------------------------------------------------------
__global__ void final_fuse(const float4* __restrict__ c2, const float4* __restrict__ sk,
                           const float* __restrict__ st_b, const float* __restrict__ st_s,
                           const float* __restrict__ gb, const float* __restrict__ bb,
                           const float* __restrict__ gs, const float* __restrict__ bs,
                           float4* __restrict__ out, int NPTS)
{
    size_t i = (size_t)blockIdx.x * 256 + threadIdx.x;
    size_t total4 = (size_t)128 * (size_t)NPTS / 4;
    if (i >= total4) return;
    int o = (int)(i / (size_t)(NPTS / 4));
    float rb = st_b[o + 128], rs = st_s[o + 128];
    float base = bb[o] + bs[o] - gb[o] * st_b[o] * rb - gs[o] * st_s[o] * rs;
    float g2 = gb[o] * rb, g1 = gs[o] * rs;
    float4 a = c2[i], b = sk[i], r;
    r.x = fmaxf(g2 * a.x + g1 * b.x + base, 0.f);
    r.y = fmaxf(g2 * a.y + g1 * b.y + base, 0.f);
    r.z = fmaxf(g2 * a.z + g1 * b.z + base, 0.f);
    r.w = fmaxf(g2 * a.w + g1 * b.w + base, 0.f);
    out[i] = r;
}

// ---------------------------------------------------------------------------
// Launcher
// ---------------------------------------------------------------------------
extern "C" void kernel_launch(void* const* d_in, const int* in_sizes, int n_in,
                              void* d_out, int out_size, void* d_ws, size_t ws_size,
                              hipStream_t stream)
{
    const float* data    = (const float*)d_in[0];   // [64][N]
    const int*   neigh   = (const int*)  d_in[1];   // [N][27]
    const float* w_a     = (const float*)d_in[2];   // [128][64][27]
    const float* w_b     = (const float*)d_in[3];   // [128][128][27]
    const float* w_skip  = (const float*)d_in[4];   // [128][64]
    const float* gamma_a = (const float*)d_in[5];
    const float* beta_a  = (const float*)d_in[6];
    const float* gamma_b = (const float*)d_in[7];
    const float* beta_b  = (const float*)d_in[8];
    const float* gamma_s = (const float*)d_in[9];
    const float* beta_s  = (const float*)d_in[10];

    const int N = in_sizes[1] / 27;                 // 65536

    // Workspace layout
    char* ws = (char*)d_ws;
    __bf16* xT       = (__bf16*)(ws + 0);                       //  8 MB  [N][64]
    __bf16* c1T      = (__bf16*)(ws + (size_t)8  * 1024 * 1024);// 16 MB  [N][128]
    float*  a_raw    = (float*) (ws + (size_t)24 * 1024 * 1024);// 32 MB  [128][N] (reused as c2_raw)
    float*  skip_raw = (float*) (ws + (size_t)56 * 1024 * 1024);// 32 MB  [128][N]
    __bf16* wat      = (__bf16*)(ws + (size_t)88 * 1024 * 1024);// 432 KB [27][128][64]
    __bf16* wbt      = (__bf16*)(ws + (size_t)89 * 1024 * 1024);// 864 KB [27][128][128]
    __bf16* wst      = (__bf16*)(ws + (size_t)90 * 1024 * 1024);// 16 KB  [1][128][64]
    float*  stats_a  = (float*) (ws + (size_t)91 * 1024 * 1024);// 3 x 256 floats
    float*  stats_s  = stats_a + 256;
    float*  stats_b  = stats_a + 512;
    float*  c2_raw   = a_raw;  // alias: a_raw dead after c1T is built

    // Dynamic LDS: 2 buffers x (weight tile + activation tile), rows padded by 8 bf16
    const size_t smemA = 2 * sizeof(__bf16) * (size_t)(128 + 128) * (64 + 8);   //  72 KB (CIN=64)
    const size_t smemB = 2 * sizeof(__bf16) * (size_t)(128 + 128) * (128 + 8);  // 136 KB (CIN=128)

    // 1) xT = transpose(bf16(x))
    transpose_to_bf16<<<dim3(N / 32, 2), dim3(32, 8), 0, stream>>>(data, xT, N, 64);

    // 2) weight repacks
    wtrans<<<(128 * 64 * 27 + 255) / 256, 256, 0, stream>>>(w_a, wat, 64, 27);
    wtrans<<<(128 * 128 * 27 + 255) / 256, 256, 0, stream>>>(w_b, wbt, 128, 27);
    wtrans<<<(128 * 64 * 1 + 255) / 256, 256, 0, stream>>>(w_skip, wst, 64, 1);

    // 3) conv_a (gather GEMM) and skip GEMM
    conv_wmma<64, 27, true ><<<N / 128, 256, smemA, stream>>>(xT, wat, neigh, a_raw, N);
    conv_wmma<64, 1,  false><<<N / 128, 256, smemA, stream>>>(xT, wst, nullptr, skip_raw, N);

    // 4) BN stats (deterministic)
    bn_stats<<<128, 256, 0, stream>>>(a_raw,    stats_a, N);
    bn_stats<<<128, 256, 0, stream>>>(skip_raw, stats_s, N);

    // 5) c1T = bf16(relu(bn_a(a_raw)))^T
    bnrelu_transpose<<<dim3(N / 32, 4), dim3(32, 8), 0, stream>>>(a_raw, stats_a, gamma_a, beta_a, c1T, N);

    // 6) conv_b (gather GEMM, Cin=128)
    conv_wmma<128, 27, true><<<N / 128, 256, smemB, stream>>>(c1T, wbt, neigh, c2_raw, N);

    // 7) BN stats for c2, then fused epilogue
    bn_stats<<<128, 256, 0, stream>>>(c2_raw, stats_b, N);
    final_fuse<<<(int)(((size_t)128 * N / 4 + 255) / 256), 256, 0, stream>>>(
        (const float4*)c2_raw, (const float4*)skip_raw,
        stats_b, stats_s, gamma_b, beta_b, gamma_s, beta_s,
        (float4*)d_out, N);
}
